// GRU_15625091023479
// MI455X (gfx1250) — matrix operand order repaired and checked by reading, not compiled, not run on hardware
//
#include <hip/hip_runtime.h>
#include <hip/hip_bf16.h>

// Problem sizes (fixed by the reference)
#define Bb 32
#define Ss 2048
#define Dd 512
#define Hh 512
#define Mm (Bb * Ss)

typedef __attribute__((ext_vector_type(16))) __bf16   v16bf;
typedef __attribute__((ext_vector_type(8)))  float    v8f;
typedef __attribute__((ext_vector_type(4)))  unsigned uvec4;
typedef __attribute__((ext_vector_type(4)))  unsigned u32x4;
typedef __attribute__((ext_vector_type(8)))  int      i32x8;
typedef __attribute__((ext_vector_type(4)))  int      i32x4;

union FragU { uvec4 u[2]; v16bf v; };

// Build a 16x bf16 WMMA operand fragment from two 16-byte chunks.
__device__ __forceinline__ v16bf load_frag(const __bf16* p0, const __bf16* p1) {
  FragU f;
  f.u[0] = *reinterpret_cast<const uvec4*>(p0);
  f.u[1] = *reinterpret_cast<const uvec4*>(p1);
  return f.v;
}

// ---------------------------------------------------------------------------
// Tensor Data Mover: 2D tile load (global -> LDS) with optional LDS padding.
// Packs D# group0/group1 per CDNA5 ISA ch.8; groups 2/3 zero (2D tensor).
// This toolchain exposes the 6-arg builtin:
//   (u32x4 g0, i32x8 g1, i32x4 g2, i32x4 g3, i32x8 extra, i32 cpol)
// ---------------------------------------------------------------------------
#define HAVE_TDM __has_builtin(__builtin_amdgcn_tensor_load_to_lds)

#if HAVE_TDM
__device__ __forceinline__ void tdm_load_2d(unsigned lds_addr, const void* gptr,
                                            unsigned rows, unsigned row_elems,
                                            unsigned src_stride_elems,
                                            unsigned pad_interval_code,
                                            unsigned pad_amount_code,
                                            bool pad_en) {
  unsigned long long ga = (unsigned long long)(uintptr_t)gptr;
  u32x4 g0;
  g0[0] = 1u;                                        // count=1, user descriptor
  g0[1] = lds_addr;                                  // LDS byte address
  g0[2] = (unsigned)(ga & 0xFFFFFFFFu);              // global_addr[31:0]
  g0[3] = (unsigned)((ga >> 32) & 0x1FFFFFFu)        // global_addr[56:32]
        | (2u << 30);                                // type = 2 ("image")
  unsigned w0 = (1u << 16);                          // data_size=1 (2B bf16)
  if (pad_en)
    w0 |= (1u << 20) | (pad_interval_code << 22) | (pad_amount_code << 25);
  i32x8 g1;
  g1[0] = (int)w0;
  g1[1] = (int)((row_elems & 0xFFFFu) << 16);                          // tensor_dim0 lo
  g1[2] = (int)(((row_elems >> 16) & 0xFFFFu) | ((rows & 0xFFFFu) << 16)); // dim0 hi | dim1 lo
  g1[3] = (int)(((rows >> 16) & 0xFFFFu) | ((row_elems & 0xFFFFu) << 16)); // dim1 hi | tile_dim0
  g1[4] = (int)(rows & 0xFFFFu);                                       // tile_dim1 | tile_dim2=0
  g1[5] = (int)src_stride_elems;                                       // dim0_stride lo32
  g1[6] = 0;                                                           // stride hi | dim1_stride lo
  g1[7] = 0;
  i32x4 z4 = {0, 0, 0, 0};
  i32x8 z8 = {0, 0, 0, 0, 0, 0, 0, 0};
  __builtin_amdgcn_tensor_load_to_lds(g0, g1, z4, z4, z8, 0);
}
#endif

// ---------------------------------------------------------------------------
// Kernel 1: convert W1 and W3 to bf16 (both are H*D = H*H = 262144 elements)
// ---------------------------------------------------------------------------
__global__ __launch_bounds__(256) void prep_bf16(const float* __restrict__ W1,
                                                 const float* __restrict__ W3,
                                                 __bf16* __restrict__ W1b,
                                                 __bf16* __restrict__ W3b) {
  int i = blockIdx.x * blockDim.x + threadIdx.x;
  W1b[i] = (__bf16)W1[i];
  W3b[i] = (__bf16)W3[i];
}

// ---------------------------------------------------------------------------
// Kernel 2: cadd[b,h] = x@W2^T + b1 + b2 + b3   (tiny: 16384 outputs, K=512)
// ---------------------------------------------------------------------------
__global__ __launch_bounds__(256) void cadd_kernel(const float* __restrict__ x,
                                                   const float* __restrict__ W2,
                                                   const float* __restrict__ b1,
                                                   const float* __restrict__ b2,
                                                   const float* __restrict__ b3,
                                                   float* __restrict__ cadd) {
  int id = blockIdx.x * blockDim.x + threadIdx.x;   // 0..16383
  int b = id >> 9, h = id & (Hh - 1);
  float acc = b1[h] + b2[h] + b3[h];
  const float* xr = x + (size_t)b * Dd;
  const float* wr = W2 + (size_t)h * Dd;
#pragma unroll 8
  for (int d = 0; d < Dd; ++d) acc += xr[d] * wr[d];
  cadd[id] = acc;
}

// ---------------------------------------------------------------------------
// Kernel 3: pre[r,n] = (y@W1^T)[r,n] + cadd[r>>11, n]   written into d_out
//   M=65536, N=512, K=512.  WG tile 64x64, 4 waves, each wave 32x32 (2x2 WMMA)
// ---------------------------------------------------------------------------
#define LDA_P 72   // padded LDS strides (bf16 elems) to avoid bank conflicts
#define LDB_P 72   // 64-elem rows (32 dwords -> interval code 4), 16B pad (code 3)

__global__ __launch_bounds__(128) void gemm_pre(const float* __restrict__ y,
                                                const __bf16* __restrict__ W1b,
                                                const float* __restrict__ cadd,
                                                float* __restrict__ pre) {
  __shared__ __bf16 As[64 * LDA_P];
  __shared__ __bf16 Bs[64 * LDB_P];

  const int tid   = threadIdx.x;
  const int wave  = tid >> 5, lane = tid & 31;
  const int lrow  = lane & 15, lhalf = lane >> 4;
  const int wm    = wave >> 1, wn = wave & 1;        // 2x2 wave grid
  const int gm0   = blockIdx.y * 64;
  const int gn0   = blockIdx.x * 64;

  v8f acc[2][2] = {};

  for (int kc = 0; kc < Dd; kc += 64) {
    // Stage B tile (bf16 64x64) with the Tensor Data Mover, overlapped with
    // the fp32->bf16 A staging below.
#if HAVE_TDM
    if (wave == 0) {
      tdm_load_2d((unsigned)(uintptr_t)Bs, W1b + (size_t)gn0 * Dd + kc,
                  /*rows=*/64, /*row_elems=*/64, /*stride=*/Dd,
                  /*interval=*/4, /*amount=*/3, /*pad=*/true);
    }
#else
#pragma unroll
    for (int it = 0; it < 4; ++it) {
      int u = it * 128 + tid;
      int r = u >> 3, c8 = u & 7;
      *reinterpret_cast<uvec4*>(Bs + r * LDB_P + c8 * 8) =
          *reinterpret_cast<const uvec4*>(W1b + (size_t)(gn0 + r) * Dd + kc + c8 * 8);
    }
#endif
    // Stage A tile (fp32 -> bf16): 64x64, 8 float4 loads per thread
#pragma unroll
    for (int it = 0; it < 8; ++it) {
      int e = it * 512 + tid * 4;
      int r = e >> 6, c = e & 63;
      const float4 v = *reinterpret_cast<const float4*>(
          y + (size_t)(gm0 + r) * Dd + kc + c);
      __bf16* dst = As + r * LDA_P + c;
      dst[0] = (__bf16)v.x; dst[1] = (__bf16)v.y;
      dst[2] = (__bf16)v.z; dst[3] = (__bf16)v.w;
    }
#if HAVE_TDM
    if (wave == 0) __builtin_amdgcn_s_wait_tensorcnt(0);
#endif
    __syncthreads();

#pragma unroll
    for (int ks = 0; ks < 64; ks += 32) {
      v16bf af[2], bfg[2];
#pragma unroll
      for (int mi = 0; mi < 2; ++mi) {
        const __bf16* p = As + (wm * 32 + mi * 16 + lrow) * LDA_P + ks + lhalf * 8;
        af[mi] = load_frag(p, p + 16);            // A: K chunks at k, k+16
      }
#pragma unroll
      for (int ni = 0; ni < 2; ++ni) {
        const __bf16* p = Bs + (wn * 32 + ni * 16 + lrow) * LDB_P + ks + lhalf * 8;
        bfg[ni] = load_frag(p, p + 16);           // B: K chunks at k, k+16
      }
#pragma unroll
      for (int mi = 0; mi < 2; ++mi)
#pragma unroll
        for (int ni = 0; ni < 2; ++ni)
          acc[mi][ni] = __builtin_amdgcn_wmma_f32_16x16x32_bf16(
              false, af[mi], false, bfg[ni], (short)0, acc[mi][ni], false, false);
    }
    __syncthreads();
  }

  // Epilogue: C/D layout -> row = base + 8*lhalf + i, col = base_n + lrow
#pragma unroll
  for (int mi = 0; mi < 2; ++mi) {
#pragma unroll
    for (int ni = 0; ni < 2; ++ni) {
      const int n     = gn0 + wn * 32 + ni * 16 + lrow;
      const int mbase = gm0 + wm * 32 + mi * 16 + lhalf * 8;
#pragma unroll
      for (int i = 0; i < 8; ++i) {
        int r = mbase + i;
        int b = r >> 11;                           // S = 2048 = 2^11
        pre[(size_t)r * Hh + n] = acc[mi][ni][i] + cadd[b * Hh + n];
      }
    }
  }
}

// ---------------------------------------------------------------------------
// Kernel 4: persistent scan.  8 WGs, each owns 64 columns; W3 slice + h tile
// in LDS; grid-wide barrier between the 2048 recurrence steps.
// ---------------------------------------------------------------------------
#define SCAN_WGS 8
#define W3_P 520   // 512-elem rows (256 dwords -> interval code 7), 16B pad (code 3)
#define HS_P 520

__device__ __forceinline__ void grid_sync(unsigned* cnt, unsigned nwg) {
  __syncthreads();
  if (threadIdx.x == 0) {
    unsigned* gen = cnt + 1;
    unsigned g = __hip_atomic_load(gen, __ATOMIC_ACQUIRE, __HIP_MEMORY_SCOPE_AGENT);
    unsigned a = __hip_atomic_fetch_add(cnt, 1u, __ATOMIC_ACQ_REL, __HIP_MEMORY_SCOPE_AGENT);
    if (a == nwg - 1u) {
      __hip_atomic_store(cnt, 0u, __ATOMIC_RELAXED, __HIP_MEMORY_SCOPE_AGENT);
      __hip_atomic_fetch_add(gen, 1u, __ATOMIC_RELEASE, __HIP_MEMORY_SCOPE_AGENT);
    } else {
      while (__hip_atomic_load(gen, __ATOMIC_ACQUIRE, __HIP_MEMORY_SCOPE_AGENT) == g) {
        __builtin_amdgcn_s_sleep(2);
      }
    }
  }
  __syncthreads();
}

__global__ __launch_bounds__(256) void scan_kernel(float* __restrict__ pre_out,
                                                   float* __restrict__ hlast,
                                                   const float* __restrict__ hx,
                                                   const __bf16* __restrict__ W3b,
                                                   __bf16* __restrict__ hbuf0,
                                                   __bf16* __restrict__ hbuf1,
                                                   unsigned* __restrict__ syncp) {
  extern __shared__ char smem[];
  __bf16* W3s = reinterpret_cast<__bf16*>(smem);   // 64 x W3_P  (~65 KB)
  __bf16* Hs  = W3s + 64 * W3_P;                   // 32 x HS_P  (~33 KB)

  const int tid  = threadIdx.x;
  const int wg   = blockIdx.x;                     // 0..7
  const int n0   = wg * 64;
  const int wave = tid >> 5, lane = tid & 31;
  const int lrow = lane & 15, lhalf = lane >> 4;
  const int mt   = wave >> 2;                      // batch half: rows 0-15 / 16-31
  const int nt   = wave & 3;                       // 16-col sub-tile

  // Stage this WG's W3 slice (rows n0..n0+63, all K=512) into LDS once.
#if HAVE_TDM
  if (wave == 0) {
    tdm_load_2d((unsigned)(uintptr_t)W3s, W3b + (size_t)n0 * Hh,
                /*rows=*/64, /*row_elems=*/512, /*stride=*/Hh,
                /*interval=*/7, /*amount=*/3, /*pad=*/true);
    __builtin_amdgcn_s_wait_tensorcnt(0);
  }
#else
#pragma unroll
  for (int it = 0; it < 16; ++it) {
    int u = it * 256 + tid;
    int r = u >> 6, c = u & 63;
    *reinterpret_cast<uvec4*>(W3s + r * W3_P + c * 8) =
        *reinterpret_cast<const uvec4*>(W3b + (size_t)(n0 + r) * Hh + c * 8);
  }
#endif
  // Initialize h0 (bf16) for our 64 columns.
#pragma unroll
  for (int it = 0; it < 8; ++it) {
    int e = it * 256 + tid;
    int r = e >> 6, c = e & 63;
    hbuf0[r * Hh + n0 + c] = (__bf16)hx[r * Hh + n0 + c];
  }
  __threadfence();
  grid_sync(syncp, SCAN_WGS);

  const __bf16* hcur = hbuf0;
  __bf16*       hnxt = hbuf1;
  const unsigned hs_lds = (unsigned)(uintptr_t)Hs;

  for (int t = 0; t < Ss; ++t) {
    // Stage full h (32x512 bf16) into LDS via TDM.
#if HAVE_TDM
    if (wave == 0) {
      tdm_load_2d(hs_lds, hcur, /*rows=*/32, /*row_elems=*/512, /*stride=*/Hh,
                  /*interval=*/7, /*amount=*/3, /*pad=*/true);
      __builtin_amdgcn_s_wait_tensorcnt(0);
    }
#else
#pragma unroll
    for (int it = 0; it < 8; ++it) {
      int u = it * 256 + tid;
      int r = u >> 6, c = u & 63;
      *reinterpret_cast<uvec4*>(Hs + r * HS_P + c * 8) =
          *reinterpret_cast<const uvec4*>(hcur + r * Hh + c * 8);
    }
#endif
    __syncthreads();

    v8f acc = {};
#pragma unroll 4
    for (int k0 = 0; k0 < Hh; k0 += 32) {
      const __bf16* pa = Hs + (mt * 16 + lrow) * HS_P + k0 + lhalf * 8;
      v16bf af = load_frag(pa, pa + 16);
      const __bf16* pb = W3s + (nt * 16 + lrow) * W3_P + k0 + lhalf * 8;
      v16bf bf = load_frag(pb, pb + 16);
      acc = __builtin_amdgcn_wmma_f32_16x16x32_bf16(
          false, af, false, bf, (short)0, acc, false, false);
    }

    const int n     = n0 + nt * 16 + lrow;
    const int bbase = mt * 16 + lhalf * 8;
#pragma unroll
    for (int i = 0; i < 8; ++i) {
      int b = bbase + i;
      size_t idx = ((size_t)b * Ss + t) * Hh + n;
      float v = acc[i] + pre_out[idx];             // pre was staged in-place
      float h = 1.0f / (1.0f + __expf(-v));
      pre_out[idx]     = h;                        // overwrite pre with output
      hnxt[b * Hh + n] = (__bf16)h;
      if (t == Ss - 1) hlast[b * Hh + n] = h;
      // Prefetch next step's pre values; overlaps HBM latency with the
      // grid barrier below (global_prefetch_b8, no LOADcnt cost).
      if (t + 1 < Ss) __builtin_prefetch(pre_out + idx + Hh, 0, 1);
    }
    __threadfence();
    grid_sync(syncp, SCAN_WGS);
    __bf16* tmp = hnxt; hnxt = const_cast<__bf16*>(hcur); hcur = tmp;
  }
}

// ---------------------------------------------------------------------------
// Host entry
// ---------------------------------------------------------------------------
extern "C" void kernel_launch(void* const* d_in, const int* in_sizes, int n_in,
                              void* d_out, int out_size, void* d_ws, size_t ws_size,
                              hipStream_t stream) {
  (void)in_sizes; (void)n_in; (void)out_size; (void)ws_size;
  const float* y  = (const float*)d_in[0];
  const float* x  = (const float*)d_in[1];
  const float* hx = (const float*)d_in[2];
  const float* W1 = (const float*)d_in[3];
  const float* b1 = (const float*)d_in[4];
  const float* W2 = (const float*)d_in[5];
  const float* b2 = (const float*)d_in[6];
  const float* W3 = (const float*)d_in[7];
  const float* b3 = (const float*)d_in[8];

  float* out   = (float*)d_out;                       // [B,S,H] output
  float* hlast = out + (size_t)Bb * Ss * Hh;          // then h_last [B,H]

  // Workspace layout (~1.2 MB)
  char* ws = (char*)d_ws;
  __bf16*   W1b   = (__bf16*)(ws);                    // 512 KB
  __bf16*   W3b   = (__bf16*)(ws + 524288);           // 512 KB
  float*    cadd  = (float*)(ws + 1048576);           // 64 KB
  __bf16*   hb0   = (__bf16*)(ws + 1114112);          // 32 KB
  __bf16*   hb1   = (__bf16*)(ws + 1146880);          // 32 KB
  unsigned* syncp = (unsigned*)(ws + 1179648);        // barrier state

  (void)hipMemsetAsync(syncp, 0, 256, stream);        // deterministic barrier init

  prep_bf16<<<1024, 256, 0, stream>>>(W1, W3, W1b, W3b);
  cadd_kernel<<<64, 256, 0, stream>>>(x, W2, b1, b2, b3, cadd);
  gemm_pre<<<dim3(Hh / 64, Mm / 64), 128, 0, stream>>>(y, W1b, cadd, out);

  size_t smem = (size_t)(64 * W3_P + 32 * HS_P) * sizeof(__bf16);  // ~98 KB
  (void)hipFuncSetAttribute((const void*)scan_kernel,
                            hipFuncAttributeMaxDynamicSharedMemorySize, (int)smem);
  scan_kernel<<<SCAN_WGS, 256, smem, stream>>>(out, hlast, hx, W3b, hb0, hb1, syncp);
}
